// RIMCell_1236950582121
// MI455X (gfx1250) — matrix operand order repaired and checked by reading, not compiled, not run on hardware
//
#include <hip/hip_runtime.h>

// ---------------- problem constants ----------------
#define B_   256
#define U_   24
#define H_   512
#define D_   512
#define IK_  64
#define IV_  400
#define CK_  32
#define NCH_ 4
#define CV_  512
#define KTOP 8

typedef __attribute__((ext_vector_type(16))) __bf16       v16bf;
typedef __attribute__((ext_vector_type(8)))  float        v8f;
typedef __attribute__((ext_vector_type(8)))  unsigned int v8u;

// round-to-nearest-even f32 -> bf16, packed pair into one dword
__device__ __forceinline__ unsigned int pack2_bf16(float a, float b) {
  unsigned int ua = __float_as_uint(a);
  unsigned int ub = __float_as_uint(b);
  ua += 0x7FFFu + ((ua >> 16) & 1u);
  ub += 0x7FFFu + ((ub >> 16) & 1u);
  return (ua >> 16) | (ub & 0xFFFF0000u);
}

// LDS tile layout: [row][k], 32 bf16 per row packed as 16 dwords, stride 17
// dwords (padding -> conflict-free across 16 lanes).

// A-operand fragment (16x32 bf16): lane m = lane&15, half = lane>>4.
// VGPR j holds K = (j/4)*16 + half*8 + (j%4)*2 (ISA 7.12.2 16-bit A layout).
__device__ __forceinline__ v16bf load_frag_a(const unsigned int* tile, int row0, int lane) {
  const unsigned int* p = tile + (row0 + (lane & 15)) * 17;
  int half4 = ((lane >> 4) & 1) * 4;
  v8u d;
#pragma unroll
  for (int j = 0; j < 8; ++j) {
    int dw = ((j >> 2) << 3) + half4 + (j & 3);   // (k>>1)
    d[j] = p[dw];
  }
  return __builtin_bit_cast(v16bf, d);
}

// B-operand fragment (32x16 bf16 = columns of W): lane n = lane&15.
// VGPR j holds K = (j/4)*16 + half*8 + (j%4)*2 (same order as A).
__device__ __forceinline__ v16bf load_frag_b(const unsigned int* tile, int col0, int lane) {
  const unsigned int* p = tile + (col0 + (lane & 15)) * 17;
  int half4 = ((lane >> 4) & 1) * 4;
  v8u d;
#pragma unroll
  for (int j = 0; j < 8; ++j) {
    d[j] = p[((j >> 2) << 3) + half4 + (j & 3)];  // (k>>1)
  }
  return __builtin_bit_cast(v16bf, d);
}

// ---------------------------------------------------------------------------
// Generic per-u batched GEMM: C[row, u, n] (op)= sum_k A[row, u, k] * W[u, k, n]
// A: f32, strided rows; W: f32 row-major (K x N) per u; bf16 WMMA, f32 accum.
// mode 0: C = acc (+bias)     mode 1: C += acc
// mode 2: C = mask*(acc + h_add) + (1-mask)*hs_in      (final hs blend)
// ---------------------------------------------------------------------------
template <int BN>
__global__ __launch_bounds__(256) void gemm_bf16(
    const float* __restrict__ A, long a_rstride, long a_ustride,
    const float* __restrict__ W, long w_ustride,
    float* __restrict__ C, long c_rstride, long c_ustride,
    int M, int N, int K,
    const float* __restrict__ bias, int mode,
    const float* __restrict__ h_add, const float* __restrict__ hs_in,
    const float* __restrict__ mask) {
  (void)M;
  constexpr int BM = 128, BK = 32;
  constexpr int WN = BN / 32;     // waves along N (2 or 4)
  constexpr int WM = 8 / WN;      // waves along M (4 or 2)
  constexpr int TM = BM / WM;     // rows per wave (32 or 64)
  constexpr int MF = TM / 16;     // m-fragments per wave (2 or 4)
  constexpr int NPAR   = 256 / BN;
  constexpr int PASSES = 8 / NPAR;

  __shared__ unsigned int As[BM * 17];
  __shared__ unsigned int Ws[BN * 17];

  const int u    = blockIdx.z;
  const int bm   = blockIdx.y * BM;
  const int bn   = blockIdx.x * BN;
  const int tid  = threadIdx.x;
  const int lane = tid & 31;
  const int wave = tid >> 5;
  const int wn   = wave % WN;
  const int wm   = wave / WN;

  const float* Ap = A + (long)u * a_ustride;
  const float* Wp = W + (long)u * w_ustride;

  // W-staging thread mapping (fixed across k-steps)
  const int wsn  = tid % BN;                        // column within tile
  const int wkq0 = tid / BN;                        // k-quad base
  // clamp: OOB columns read column N-1; their outputs are discarded in the
  // epilogue (col >= N), so no guard is needed in the hot loop.
  const int wng  = (bn + wsn < N) ? (bn + wsn) : (N - 1);

  v8f acc[MF][2];
#pragma unroll
  for (int i = 0; i < MF; ++i) {
    acc[i][0] = (v8f)0.f;
    acc[i][1] = (v8f)0.f;
  }

  const int ksteps = (K + BK - 1) / BK;
  for (int ks = 0; ks < ksteps; ++ks) {
    const int k0 = ks * BK;

    if (k0 + BK <= K) {
      // ======== branch-free hot path (block-uniform) ========
      // A tile: 128 x 32 f32 -> bf16
#pragma unroll
      for (int it = 0; it < 4; ++it) {
        int idx = it * 256 + tid;
        int r   = idx >> 3;
        int kq  = idx & 7;
        const float4 v = *reinterpret_cast<const float4*>(
            Ap + (long)(bm + r) * a_rstride + k0 + kq * 4);
        unsigned int* dst = &As[r * 17 + kq * 2];
        dst[0] = pack2_bf16(v.x, v.y);
        dst[1] = pack2_bf16(v.z, v.w);
      }
      // W tile: 32 x BN f32 -> bf16, transposed to [n][k]
#pragma unroll
      for (int p = 0; p < PASSES; ++p) {
        int kq = p * NPAR + wkq0;
        const float* src = Wp + (long)(k0 + kq * 4) * N + wng;
        float va = src[0];
        float vb = src[(long)N];
        float vc = src[2L * N];
        float vd = src[3L * N];
        unsigned int* dst = &Ws[wsn * 17 + kq * 2];
        dst[0] = pack2_bf16(va, vb);
        dst[1] = pack2_bf16(vc, vd);
      }
    } else {
      // ======== guarded tail (runs at most once, K=400 case) ========
#pragma unroll
      for (int it = 0; it < 4; ++it) {
        int idx = it * 256 + tid;
        int r   = idx >> 3;
        int kq  = idx & 7;
        int kg  = k0 + kq * 4;
        const float* src = Ap + (long)(bm + r) * a_rstride;
        float4 v = make_float4(0.f, 0.f, 0.f, 0.f);
        if (kg + 3 < K) {
          v = *reinterpret_cast<const float4*>(src + kg);
        } else {
          if (kg + 0 < K) v.x = src[kg + 0];
          if (kg + 1 < K) v.y = src[kg + 1];
          if (kg + 2 < K) v.z = src[kg + 2];
          if (kg + 3 < K) v.w = src[kg + 3];
        }
        unsigned int* dst = &As[r * 17 + kq * 2];
        dst[0] = pack2_bf16(v.x, v.y);
        dst[1] = pack2_bf16(v.z, v.w);
      }
#pragma unroll
      for (int p = 0; p < PASSES; ++p) {
        int kq = p * NPAR + wkq0;
        int kg = k0 + kq * 4;
        const float* src = Wp + (long)kg * N + wng;
        float va = (kg + 0 < K) ? src[0] : 0.f;
        float vb = (kg + 1 < K) ? src[(long)N] : 0.f;
        float vc = (kg + 2 < K) ? src[2L * N] : 0.f;
        float vd = (kg + 3 < K) ? src[3L * N] : 0.f;
        unsigned int* dst = &Ws[wsn * 17 + kq * 2];
        dst[0] = pack2_bf16(va, vb);
        dst[1] = pack2_bf16(vc, vd);
      }
    }

    if (ks + 1 < ksteps) {  // pull next K-tile toward L2 (global_prefetch_b8)
      __builtin_prefetch(Ap + (long)(bm + (tid >> 1)) * a_rstride + k0 + BK, 0, 1);
      __builtin_prefetch(Wp + (long)(k0 + BK + (tid & 31)) * N + bn, 0, 1);
    }

    __syncthreads();

    // ---- WMMA compute ----
    v16bf bf0 = load_frag_b(Ws, wn * 32, lane);
    v16bf bf1 = load_frag_b(Ws, wn * 32 + 16, lane);
#pragma unroll
    for (int i = 0; i < MF; ++i) {
      v16bf af = load_frag_a(As, wm * TM + i * 16, lane);
      acc[i][0] = __builtin_amdgcn_wmma_f32_16x16x32_bf16(
          false, af, false, bf0, (short)0, acc[i][0], false, false);
      acc[i][1] = __builtin_amdgcn_wmma_f32_16x16x32_bf16(
          false, af, false, bf1, (short)0, acc[i][1], false, false);
    }
    __syncthreads();
  }

  // ---- epilogue ----
  const int half = (lane >> 4) & 1;
  const int nl   = lane & 15;
#pragma unroll
  for (int i = 0; i < MF; ++i) {
#pragma unroll
    for (int j = 0; j < 2; ++j) {
      int col = bn + wn * 32 + j * 16 + nl;
      if (col >= N) continue;
      float bv = bias ? bias[col] : 0.f;
#pragma unroll
      for (int r = 0; r < 8; ++r) {
        int  row  = bm + wm * TM + i * 16 + half * 8 + r;
        long coff = (long)row * c_rstride + (long)u * c_ustride + col;
        float val = acc[i][j][r] + bv;
        if (mode == 0) {
          C[coff] = val;
        } else if (mode == 1) {
          C[coff] += val;
        } else {
          float m = mask[row * U_ + u];
          C[coff] = m * (val + h_add[coff]) + (1.f - m) * hs_in[coff];
        }
      }
    }
  }
}

// ---------------------------------------------------------------------------
// scores -> top-K mask -> 2-way softmax -> masked inp. One block per batch row.
// ---------------------------------------------------------------------------
__global__ __launch_bounds__(256) void score_mask_inp_kernel(
    const float* __restrict__ q,        // (B,U,IK)
    const float* __restrict__ k0,       // (B,IK)
    const float* __restrict__ key_b,    // (IK)  == k row 1
    const float* __restrict__ v0,       // (B,IV)
    const float* __restrict__ value_b,  // (IV)  == v row 1
    float* __restrict__ mask_out,       // (B,U)
    float* __restrict__ inp) {          // (B,U,IV)
  int b   = blockIdx.x;
  int tid = threadIdx.x;
  __shared__ float s_k0[IK_], s_k1[IK_], s_v0[IV_], s_v1[IV_];
  __shared__ float s0[U_], s1[U_], p0[U_], p1[U_], msk[U_];

  if (tid < IK_) {
    s_k0[tid] = k0[b * IK_ + tid];
    s_k1[tid] = key_b[tid];
  }
  for (int v = tid; v < IV_; v += 256) {
    s_v0[v] = v0[b * IV_ + v];
    s_v1[v] = value_b[v];
  }
  __syncthreads();

  int wave = tid >> 5, lane = tid & 31;
  for (int u = wave; u < U_; u += 8) {
    const float* qp = q + ((long)b * U_ + u) * IK_;
    float d0 = 0.f, d1 = 0.f;
    for (int i = lane; i < IK_; i += 32) {
      float qv = qp[i];
      d0 += qv * s_k0[i];
      d1 += qv * s_k1[i];
    }
#pragma unroll
    for (int off = 16; off > 0; off >>= 1) {  // wave32 reduction
      d0 += __shfl_xor(d0, off, 32);
      d1 += __shfl_xor(d1, off, 32);
    }
    if (lane == 0) {
      s0[u] = d0 * 0.125f;  // 1/sqrt(IK)
      s1[u] = d1 * 0.125f;
    }
  }
  __syncthreads();

  if (tid < U_) {
    float mys = s0[tid];
    int rank = 0;
    for (int j = 0; j < U_; ++j) {
      float sj = s0[j];
      rank += (sj > mys || (sj == mys && j < tid)) ? 1 : 0;
    }
    float m = (rank < KTOP) ? 1.f : 0.f;
    msk[tid] = m;
    mask_out[b * U_ + tid] = m;
    float a = s0[tid], c = s1[tid];
    float mx = fmaxf(a, c);
    float e0 = __expf(a - mx), e1 = __expf(c - mx);
    float inv = 1.f / (e0 + e1);
    p0[tid] = e0 * inv;
    p1[tid] = e1 * inv;
  }
  __syncthreads();

  for (int idx = tid; idx < U_ * IV_; idx += 256) {
    int u = idx / IV_, v = idx - u * IV_;
    inp[((long)b * U_ + u) * IV_ + v] = msk[u] * (p0[u] * s_v0[v] + p1[u] * s_v1[v]);
  }
}

// ---------------------------------------------------------------------------
// LSTM gates: h_t to workspace, cs_new (mask-blended) straight to output.
// ---------------------------------------------------------------------------
__global__ __launch_bounds__(256) void gates_kernel(
    const float* __restrict__ preact,   // (B,U,4H)
    const float* __restrict__ cs,       // (B,U,H)
    const float* __restrict__ mask,     // (B,U)
    float* __restrict__ h_out,          // (B,U,H)
    float* __restrict__ cs_new,         // (B,U,H)
    long total) {
  long i = (long)blockIdx.x * 256 + threadIdx.x;
  if (i >= total) return;
  long bu = i >> 9;         // /H
  int  hh = (int)(i & 511); // %H
  const float* p = preact + bu * (4L * H_);
  float it = 1.f / (1.f + __expf(-p[hh]));
  float ft = 1.f / (1.f + __expf(-p[H_ + hh]));
  float ot = 1.f / (1.f + __expf(-p[2 * H_ + hh]));
  float gt = tanhf(p[3 * H_ + hh]);
  float cv = cs[i] * ft + it * gt;
  h_out[i] = ot * tanhf(cv);
  float m = mask[bu];
  cs_new[i] = m * cv + (1.f - m) * cs[i];
}

// ---------------------------------------------------------------------------
// Communication attention: one block per (head, batch). att 24x24 + ctx.
// V-slice (48 KB f32) staged with CDNA5 async global->LDS DMA (ASYNCcnt).
// ---------------------------------------------------------------------------
__global__ __launch_bounds__(256) void comm_attn_kernel(
    const float* __restrict__ qc,    // (B,U,NCH*CQ)
    const float* __restrict__ kc,    // (B,U,NCH*CK)
    const float* __restrict__ vc,    // (B,U,NCH*CV)
    const float* __restrict__ mask,  // (B,U)
    float* __restrict__ ctx) {       // (B,U,NCH*CV), [b,u,n*CV+v]
  int n = blockIdx.x;  // head
  int b = blockIdx.y;
  int tid = threadIdx.x;
  __shared__ float s_q[U_][CK_];
  __shared__ float s_k[U_][CK_];
  __shared__ float s_att[U_][U_ + 1];
  __shared__ __align__(16) float s_v[U_][CV_];  // 48 KB of the 320 KB WGP LDS

  // kick off async V-slice copy first (global_load_async_to_lds_b128),
  // then overlap the q/k staging with it.
  {
    // 24*512 floats = 3072 float4 chunks, 12 per thread
    for (int idx = tid; idx < (U_ * CV_) / 4; idx += 256) {
      int t = idx >> 7;              // /(CV_/4)
      int v = (idx & 127) << 2;      // *4
      unsigned lds_off = (unsigned)(size_t)(&s_v[t][v]);
      unsigned long long gaddr = (unsigned long long)(size_t)(
          vc + ((long)b * U_ + t) * (NCH_ * CV_) + n * CV_ + v);
      asm volatile("global_load_async_to_lds_b128 %0, %1, off"
                   :: "v"(lds_off), "v"(gaddr)
                   : "memory");
    }
  }

  for (int idx = tid; idx < U_ * CK_; idx += 256) {
    int u = idx / CK_, c = idx - u * CK_;
    s_q[u][c] = qc[((long)b * U_ + u) * (NCH_ * CK_) + n * CK_ + c];
    s_k[u][c] = kc[((long)b * U_ + u) * (NCH_ * CK_) + n * CK_ + c];
  }

  // drain this wave's async copies, then barrier so all waves' data is visible
  asm volatile("s_wait_asynccnt 0" ::: "memory");
  __syncthreads();

  const float scal = 0.17677669529663687f;  // 1/sqrt(CK)
  for (int idx = tid; idx < U_ * U_; idx += 256) {
    int u = idx / U_, t = idx - u * U_;
    float d = 0.f;
#pragma unroll
    for (int c = 0; c < CK_; ++c) d += s_q[u][c] * s_k[t][c];
    s_att[u][t] = d * scal;
  }
  __syncthreads();

  if (tid < U_) {  // row-wise softmax, then query-side mask
    float mx = -3.4e38f;
    for (int t = 0; t < U_; ++t) mx = fmaxf(mx, s_att[tid][t]);
    float sum = 0.f;
    for (int t = 0; t < U_; ++t) {
      float e = __expf(s_att[tid][t] - mx);
      s_att[tid][t] = e;
      sum += e;
    }
    float sc = mask[b * U_ + tid] / sum;
    for (int t = 0; t < U_; ++t) s_att[tid][t] *= sc;
  }
  __syncthreads();

  for (int idx = tid; idx < U_ * CV_; idx += 256) {
    int u = idx / CV_, v = idx - u * CV_;
    float d = 0.f;
#pragma unroll
    for (int t = 0; t < U_; ++t) d += s_att[u][t] * s_v[t][v];
    ctx[((long)b * U_ + u) * (NCH_ * CV_) + n * CV_ + v] = d;
  }
}

// ---------------------------------------------------------------------------
extern "C" void kernel_launch(void* const* d_in, const int* in_sizes, int n_in,
                              void* d_out, int out_size, void* d_ws, size_t ws_size,
                              hipStream_t stream) {
  (void)in_sizes; (void)n_in; (void)out_size; (void)ws_size;
  const float* x       = (const float*)d_in[0];
  const float* hs      = (const float*)d_in[1];
  const float* cs      = (const float*)d_in[2];
  const float* key_w   = (const float*)d_in[3];
  const float* key_b   = (const float*)d_in[4];
  const float* value_w = (const float*)d_in[5];
  const float* value_b = (const float*)d_in[6];
  const float* query_w = (const float*)d_in[7];
  const float* i2h_w   = (const float*)d_in[8];
  const float* h2h_w   = (const float*)d_in[9];
  const float* qc_w    = (const float*)d_in[10];
  const float* kc_w    = (const float*)d_in[11];
  const float* vc_w    = (const float*)d_in[12];
  const float* co_w    = (const float*)d_in[13];

  float* hs_out = (float*)d_out;                       // (B,U,H)
  float* cs_out = hs_out + (long)B_ * U_ * H_;         // (B,U,H)

  float* wsf = (float*)d_ws;
  size_t o = 0;
  auto take = [&](size_t nelem) { float* p = wsf + o; o += nelem; return p; };
  float* k0b   = take((size_t)B_ * IK_);
  float* v0b   = take((size_t)B_ * IV_);
  float* qb    = take((size_t)B_ * U_ * IK_);
  float* maskb = take((size_t)B_ * U_);
  float* inpb  = take((size_t)B_ * U_ * IV_);
  float* preb  = take((size_t)B_ * U_ * 4 * H_);       // later reused as ctx
  float* hb    = take((size_t)B_ * U_ * H_);
  float* qcb   = take((size_t)B_ * U_ * NCH_ * CK_);
  float* kcb   = take((size_t)B_ * U_ * NCH_ * CK_);
  float* vcb   = take((size_t)B_ * U_ * NCH_ * CV_);

  const long UH = (long)U_ * H_;

  // 1) k0 = x @ key_w + key_b            (256 x 64 x 512)
  gemm_bf16<64><<<dim3(1, 2, 1), 256, 0, stream>>>(
      x, D_, 0, key_w, 0, k0b, IK_, 0, B_, IK_, D_, key_b, 0,
      nullptr, nullptr, nullptr);
  // 2) v0 = x @ value_w + value_b        (256 x 400 x 512)
  gemm_bf16<128><<<dim3(4, 2, 1), 256, 0, stream>>>(
      x, D_, 0, value_w, 0, v0b, IV_, 0, B_, IV_, D_, value_b, 0,
      nullptr, nullptr, nullptr);
  // 3) q[:,u,:] = hs[:,u,:] @ query_w[u] (per-u 256 x 64 x 512)
  gemm_bf16<64><<<dim3(1, 2, U_), 256, 0, stream>>>(
      hs, UH, H_, query_w, (long)H_ * IK_, qb, (long)U_ * IK_, IK_,
      B_, IK_, H_, nullptr, 0, nullptr, nullptr, nullptr);
  // 4) scores -> top-8 mask -> softmax -> masked inp
  score_mask_inp_kernel<<<B_, 256, 0, stream>>>(
      qb, k0b, key_b, v0b, value_b, maskb, inpb);
  // 5) preact  = inp @ i2h_w[u]          (per-u 256 x 2048 x 400)
  gemm_bf16<128><<<dim3(16, 2, U_), 256, 0, stream>>>(
      inpb, (long)U_ * IV_, IV_, i2h_w, (long)IV_ * 4 * H_,
      preb, (long)U_ * 4 * H_, 4L * H_, B_, 4 * H_, IV_, nullptr, 0,
      nullptr, nullptr, nullptr);
  // 6) preact += hs @ h2h_w[u]           (per-u 256 x 2048 x 512)
  gemm_bf16<128><<<dim3(16, 2, U_), 256, 0, stream>>>(
      hs, UH, H_, h2h_w, (long)H_ * 4 * H_,
      preb, (long)U_ * 4 * H_, 4L * H_, B_, 4 * H_, H_, nullptr, 1,
      nullptr, nullptr, nullptr);
  // 7) gates: h_t -> ws, cs_new -> output (mask-blended)
  {
    long total = (long)B_ * U_ * H_;
    int blocks = (int)((total + 255) / 256);
    gates_kernel<<<blocks, 256, 0, stream>>>(preb, cs, maskb, hb, cs_out, total);
  }
  // 8/9) qc, kc = h @ {qc_w,kc_w}[u]     (per-u 256 x 128 x 512)
  gemm_bf16<128><<<dim3(1, 2, U_), 256, 0, stream>>>(
      hb, UH, H_, qc_w, (long)H_ * NCH_ * CK_,
      qcb, (long)U_ * NCH_ * CK_, (long)NCH_ * CK_, B_, NCH_ * CK_, H_,
      nullptr, 0, nullptr, nullptr, nullptr);
  gemm_bf16<128><<<dim3(1, 2, U_), 256, 0, stream>>>(
      hb, UH, H_, kc_w, (long)H_ * NCH_ * CK_,
      kcb, (long)U_ * NCH_ * CK_, (long)NCH_ * CK_, B_, NCH_ * CK_, H_,
      nullptr, 0, nullptr, nullptr, nullptr);
  // 10) vc = h @ vc_w[u]                 (per-u 256 x 2048 x 512)
  gemm_bf16<128><<<dim3(16, 2, U_), 256, 0, stream>>>(
      hb, UH, H_, vc_w, (long)H_ * NCH_ * CV_,
      vcb, (long)U_ * NCH_ * CV_, (long)NCH_ * CV_, B_, NCH_ * CV_, H_,
      nullptr, 0, nullptr, nullptr, nullptr);
  // 11) per-(b,head) attention, ctx written into preact buffer (reused)
  comm_attn_kernel<<<dim3(NCH_, B_), 256, 0, stream>>>(
      qcb, kcb, vcb, maskb, preb);
  // 12) hs_new = m*(ctx @ co_w[u] + h) + (1-m)*hs  -> output directly
  gemm_bf16<128><<<dim3(4, 2, U_), 256, 0, stream>>>(
      preb, (long)U_ * NCH_ * CV_, (long)NCH_ * CV_, co_w, (long)NCH_ * CV_ * H_,
      hs_out, UH, H_, B_, H_, NCH_ * CV_, nullptr, 2, hb, hs, maskb);
}
